// TransformerDecoder_33629593928311
// MI455X (gfx1250) — matrix operand order repaired and checked by reading, not compiled, not run on hardware
//
#include <hip/hip_runtime.h>
#include <hip/hip_bf16.h>

// ---------------------------------------------------------------------------
// Types for CDNA5 WMMA (gfx1250, wave32)
// ---------------------------------------------------------------------------
typedef __attribute__((ext_vector_type(16))) __bf16 v16bf;
typedef __attribute__((ext_vector_type(8)))  float  v8f;

union FragAB {
    unsigned int u[8];
    v16bf        v;
};

union Pack16 {
    uint4          q[2];
    unsigned int   w[8];
    unsigned short s[16];
};

__device__ __forceinline__ unsigned short f2bf(float f) {
    unsigned int u = __float_as_uint(f);
    unsigned int r = (u + 0x7FFFu + ((u >> 16) & 1u)) >> 16;   // RNE
    return (unsigned short)r;
}

// ---------------------------------------------------------------------------
// Block reductions (256 threads)
// ---------------------------------------------------------------------------
__device__ __forceinline__ float blk_sum(float v, float* sb) {
    const int t = threadIdx.x;
    sb[t] = v; __syncthreads();
    for (int s = 128; s > 0; s >>= 1) {
        if (t < s) sb[t] += sb[t + s];
        __syncthreads();
    }
    float r = sb[0]; __syncthreads();
    return r;
}

__device__ __forceinline__ float blk_max(float v, float* sb) {
    const int t = threadIdx.x;
    sb[t] = v; __syncthreads();
    for (int s = 128; s > 0; s >>= 1) {
        if (t < s) sb[t] = fmaxf(sb[t], sb[t + s]);
        __syncthreads();
    }
    float r = sb[0]; __syncthreads();
    return r;
}

// ---------------------------------------------------------------------------
// Elementwise fp32 -> bf16
// ---------------------------------------------------------------------------
__global__ __launch_bounds__(256)
void f32_to_bf16_kernel(const float* __restrict__ in, unsigned short* __restrict__ out, int n) {
    int i = blockIdx.x * 256 + threadIdx.x;
    if (i < n) out[i] = f2bf(in[i]);
}

// ---------------------------------------------------------------------------
// LayerNorm over D=1024, fused with the (S,B,D)->(B,S,D) swapaxes and the
// [1:] slice.  One 256-thread block per row; dual fp32/bf16 output.
// ---------------------------------------------------------------------------
template<bool WRITE_F32>
__global__ __launch_bounds__(256)
void ln_kernel(const float* __restrict__ x, const float* __restrict__ w,
               const float* __restrict__ bb, float* __restrict__ of32,
               unsigned short* __restrict__ obf, int S, int Bn) {
    __shared__ float sbuf[256];
    const int row = blockIdx.x;                 // row = b*S + s
    const int b = row / S, s = row % S;
    const size_t in_off  = ((size_t)(s + 1) * Bn + b) * 1024;
    const size_t out_off = (size_t)row * 1024;
    const int t = threadIdx.x;

    float v[4]; float sum = 0.f;
    #pragma unroll
    for (int i = 0; i < 4; ++i) { v[i] = x[in_off + t + 256 * i]; sum += v[i]; }
    sum = blk_sum(sum, sbuf);
    const float mean = sum * (1.0f / 1024.0f);
    float sq = 0.f;
    #pragma unroll
    for (int i = 0; i < 4; ++i) { float d = v[i] - mean; sq += d * d; }
    sq = blk_sum(sq, sbuf);
    const float rstd = rsqrtf(sq * (1.0f / 1024.0f) + 1e-5f);
    #pragma unroll
    for (int i = 0; i < 4; ++i) {
        const int idx = t + 256 * i;
        const float y = (v[i] - mean) * rstd * w[idx] + bb[idx];
        if (WRITE_F32) of32[out_off + idx] = y;
        obf[out_off + idx] = f2bf(y);
    }
}

// Residual add + LayerNorm, in-place fp32 (z <- LN(z+xa)) plus bf16 copy.
__global__ __launch_bounds__(256)
void resln_kernel(float* __restrict__ z, const float* __restrict__ xa,
                  const float* __restrict__ w, const float* __restrict__ bb,
                  unsigned short* __restrict__ obf) {
    __shared__ float sbuf[256];
    const size_t base = (size_t)blockIdx.x * 1024;
    const int t = threadIdx.x;
    float v[4]; float sum = 0.f;
    #pragma unroll
    for (int i = 0; i < 4; ++i) {
        const int idx = t + 256 * i;
        v[i] = z[base + idx] + xa[base + idx];
        sum += v[i];
    }
    sum = blk_sum(sum, sbuf);
    const float mean = sum * (1.0f / 1024.0f);
    float sq = 0.f;
    #pragma unroll
    for (int i = 0; i < 4; ++i) { float d = v[i] - mean; sq += d * d; }
    sq = blk_sum(sq, sbuf);
    const float rstd = rsqrtf(sq * (1.0f / 1024.0f) + 1e-5f);
    #pragma unroll
    for (int i = 0; i < 4; ++i) {
        const int idx = t + 256 * i;
        const float y = (v[i] - mean) * rstd * w[idx] + bb[idx];
        z[base + idx] = y;
        obf[base + idx] = f2bf(y);
    }
}

// Row softmax over 2048 entries (scale 1/sqrt(1024)=1/32 folded in), bf16 out.
__global__ __launch_bounds__(256)
void softmax_kernel(const float* __restrict__ sc, unsigned short* __restrict__ pr) {
    __shared__ float sbuf[256];
    const size_t base = (size_t)blockIdx.x * 2048;
    const int t = threadIdx.x;
    float v[8]; float mx = -3.0e38f;
    #pragma unroll
    for (int i = 0; i < 8; ++i) {
        v[i] = sc[base + t + 256 * i] * 0.03125f;
        mx = fmaxf(mx, v[i]);
    }
    mx = blk_max(mx, sbuf);
    float sum = 0.f;
    #pragma unroll
    for (int i = 0; i < 8; ++i) { v[i] = __expf(v[i] - mx); sum += v[i]; }
    sum = blk_sum(sum, sbuf);
    const float inv = 1.0f / sum;
    #pragma unroll
    for (int i = 0; i < 8; ++i) pr[base + t + 256 * i] = f2bf(v[i] * inv);
}

// ---------------------------------------------------------------------------
// bf16 WMMA GEMM, C(fp32) = A(MxK) * B(KxN), 128x128 tile / workgroup,
// 8 wave32 waves, each owning a 32x64 C sub-tile (2x4 WMMA frags).
// BT=true : B element (k,n) = Bm[n*ldb + k]  (weight layout W[n][k], K^T)
// BT=false: B element (k,n) = Bm[k*ldb + n]  (value layout V[k][n])
// Epilogue modes below.
// ---------------------------------------------------------------------------
enum { EP_F32 = 0, EP_BF16 = 1, EP_GELU = 2, EP_FINAL = 3 };

template<bool BT, int MODE>
__global__ __launch_bounds__(256)
void gemm_bf16(const unsigned short* __restrict__ A, long long aBatch, int lda,
               const unsigned short* __restrict__ Bm, long long bBatch, int ldb,
               void* __restrict__ OutP, long long oBatch, int ldo,
               const float* __restrict__ bias,
               const float* __restrict__ resid, long long rBatch, int ldr,
               int K) {
    // 34-short row stride = 17 dwords: coprime with 64 LDS banks, 4B aligned.
    __shared__ unsigned short As[128 * 34];
    __shared__ unsigned short Bs[128 * 34];

    const int tid  = threadIdx.x;
    const int lane = tid & 31;
    const int wv   = tid >> 5;
    const int wm   = wv >> 1;          // 0..3 : 32-row band
    const int wn   = wv & 1;           // 0..1 : 64-col band
    const int lm   = lane & 15;
    const int hl   = lane >> 4;

    const int m0 = blockIdx.y * 128;
    const int n0 = blockIdx.x * 128;

    const unsigned short* Ab = A  + (size_t)blockIdx.z * (size_t)aBatch;
    const unsigned short* Bb = Bm + (size_t)blockIdx.z * (size_t)bBatch;

    v8f acc[2][4];
    #pragma unroll
    for (int i = 0; i < 2; ++i)
        #pragma unroll
        for (int j = 0; j < 4; ++j)
            acc[i][j] = (v8f)(0.0f);

    // loader mapping: A tile 128x32, 2 threads per row (16 bf16 each)
    const int arow = tid >> 1;
    const int aseg = (tid & 1) * 16;
    // B tile loader mapping depends on layout
    const int brow = BT ? (tid >> 1) : (tid >> 3);
    const int bseg = BT ? ((tid & 1) * 16) : ((tid & 7) * 16);

    for (int kk = 0; kk < K; kk += 32) {
        {   // ---- A tile -> LDS (row-major, k contiguous) ----
            const unsigned short* ga = Ab + (size_t)(m0 + arow) * lda + kk + aseg;
            Pack16 p;
            p.q[0] = *(const uint4*)(ga);
            p.q[1] = *(const uint4*)(ga + 8);
            if (kk + 32 < K) __builtin_prefetch(ga + 32, 0, 0);
            unsigned int* dst = (unsigned int*)&As[arow * 34 + aseg];
            #pragma unroll
            for (int j = 0; j < 8; ++j) dst[j] = p.w[j];
        }
        if (BT) {   // ---- B tile (W[n][k]) -> LDS as [n][k] ----
            const unsigned short* gb = Bb + (size_t)(n0 + brow) * ldb + kk + bseg;
            Pack16 p;
            p.q[0] = *(const uint4*)(gb);
            p.q[1] = *(const uint4*)(gb + 8);
            if (kk + 32 < K) __builtin_prefetch(gb + 32, 0, 0);
            unsigned int* dst = (unsigned int*)&Bs[brow * 34 + bseg];
            #pragma unroll
            for (int j = 0; j < 8; ++j) dst[j] = p.w[j];
        } else {    // ---- B tile (V[k][n]) -> transpose into LDS [n][k] ----
            const unsigned short* gb = Bb + (size_t)(kk + brow) * ldb + n0 + bseg;
            Pack16 p;
            p.q[0] = *(const uint4*)(gb);
            p.q[1] = *(const uint4*)(gb + 8);
            if (kk + 32 < K) __builtin_prefetch(gb + (size_t)32 * ldb, 0, 0);
            #pragma unroll
            for (int j = 0; j < 16; ++j) Bs[(bseg + j) * 34 + brow] = p.s[j];
        }
        __syncthreads();

        // ---- fragments per CDNA5 ISA VGPR layouts ----
        FragAB afr[2], bfr[4];
        #pragma unroll
        for (int fm = 0; fm < 2; ++fm) {
            const int mr = wm * 32 + fm * 16 + lm;
            #pragma unroll
            for (int i = 0; i < 8; ++i) {
                // A 16x32 bf16: lanes 0-15 K={0..7,16..23}, lanes 16-31 +8
                const int k0 = ((i & 3) * 2) + ((i >> 2) * 16) + hl * 8;
                afr[fm].u[i] = *(const unsigned int*)&As[mr * 34 + k0];
            }
        }
        #pragma unroll
        for (int fn = 0; fn < 4; ++fn) {
            const int nr = wn * 64 + fn * 16 + lm;
            #pragma unroll
            for (int i = 0; i < 8; ++i) {
                // B 32x16 bf16: lanes 0-15 K={0..7,16..23}, lanes 16-31 +8
                const int k0 = ((i & 3) * 2) + ((i >> 2) * 16) + hl * 8;
                bfr[fn].u[i] = *(const unsigned int*)&Bs[nr * 34 + k0];
            }
        }
        #pragma unroll
        for (int fm = 0; fm < 2; ++fm)
            #pragma unroll
            for (int fn = 0; fn < 4; ++fn)
                acc[fm][fn] = __builtin_amdgcn_wmma_f32_16x16x32_bf16(
                    false, afr[fm].v, false, bfr[fn].v,
                    (short)0, acc[fm][fn], false, false);
        __syncthreads();
    }

    // ---- epilogue: C/D layout lanes 0-15 N=lm M=r, lanes 16-31 M=r+8 ----
    const size_t oBase = (size_t)blockIdx.z * (size_t)oBatch;
    #pragma unroll
    for (int fn = 0; fn < 4; ++fn) {
        const int gn = n0 + wn * 64 + fn * 16 + lm;
        const float bv = bias ? bias[gn] : 0.0f;
        #pragma unroll
        for (int fm = 0; fm < 2; ++fm) {
            #pragma unroll
            for (int r = 0; r < 8; ++r) {
                const int gm = m0 + wm * 32 + fm * 16 + r + hl * 8;
                const float val = acc[fm][fn][r] + bv;
                const size_t oidx = oBase + (size_t)gm * ldo + gn;
                if (MODE == EP_F32) {
                    ((float*)OutP)[oidx] = val;
                } else if (MODE == EP_BF16) {
                    ((unsigned short*)OutP)[oidx] = f2bf(val);
                } else if (MODE == EP_GELU) {
                    const float g = 0.5f * val * (1.0f + erff(val * 0.7071067811865475f));
                    ((unsigned short*)OutP)[oidx] = f2bf(g);
                } else { // EP_FINAL: out[s][b][d] = zxa + mlp
                    const float zv =
                        resid[(size_t)blockIdx.z * (size_t)rBatch + (size_t)gm * ldr + gn];
                    ((float*)OutP)[oidx] = val + zv;
                }
            }
        }
    }
}

// ---------------------------------------------------------------------------
// Host-side orchestration
// ---------------------------------------------------------------------------
extern "C" void kernel_launch(void* const* d_in, const int* in_sizes, int n_in,
                              void* d_out, int out_size, void* d_ws, size_t ws_size,
                              hipStream_t stream) {
    (void)in_sizes; (void)n_in; (void)out_size; (void)ws_size;

    const float* content = (const float*)d_in[0];
    const float* style   = (const float*)d_in[1];
    const float* ln_w    = (const float*)d_in[2];
    const float* ln_b    = (const float*)d_in[3];
    const float* Wq      = (const float*)d_in[4];
    const float* bq      = (const float*)d_in[5];
    const float* Wkv     = (const float*)d_in[6];
    const float* bkv     = (const float*)d_in[7];
    const float* Wo      = (const float*)d_in[8];
    const float* bo      = (const float*)d_in[9];
    const float* Wm1     = (const float*)d_in[10];
    const float* bm1     = (const float*)d_in[11];
    const float* Wm2     = (const float*)d_in[12];
    const float* bm2     = (const float*)d_in[13];

    const int Bn = 8, S = 2048, D = 1024, N2 = 2048;
    const size_t BSD = (size_t)Bn * S * D;          // 16,777,216 elems

    char* ws = (char*)d_ws;
    size_t off = 0;
    auto alloc = [&](size_t bytes) -> char* {
        char* p = ws + off;
        off = (off + bytes + 255) & ~(size_t)255;
        return p;
    };

    float*          zc_f32 = (float*)alloc(BSD * 4);                     // 64 MB
    unsigned short* zc_bf  = (unsigned short*)alloc(BSD * 2);            // 32 MB
    unsigned short* zs_bf  = (unsigned short*)alloc(BSD * 2);            // 32 MB (contig after zc_bf)
    unsigned short* q_bf   = (unsigned short*)alloc(BSD * 2);            // 32 MB
    unsigned short* kv_bf  = (unsigned short*)alloc((size_t)Bn * S * N2 * 2);  // 64 MB
    float*          sc_f32 = (float*)alloc((size_t)Bn * S * N2 * 4);     // 128 MB
    unsigned short* wq_bf  = (unsigned short*)alloc((size_t)D * D * 2);
    unsigned short* wkv_bf = (unsigned short*)alloc((size_t)N2 * D * 2);
    unsigned short* wo_bf  = (unsigned short*)alloc((size_t)D * D * 2);
    unsigned short* wm1_bf = (unsigned short*)alloc((size_t)D * D * 2);
    unsigned short* wm2_bf = (unsigned short*)alloc((size_t)D * D * 2);

    // Lifetime-safe aliases (dead buffers reused):
    unsigned short* probs_bf = zc_bf;   // 64 MB spans zc_bf+zs_bf (both dead post-QKV)
    unsigned short* pv_bf    = q_bf;    // q dead after scores
    float*          xa_f32   = sc_f32;  // scores dead after softmax
    unsigned short* zxa_bf   = kv_bf;   // kv dead after PV
    unsigned short* h1_bf    = q_bf;    // pv dead after Wo

    const dim3 blk(256);

    // 1) weights -> bf16
    {
        const int n = D * D;
        f32_to_bf16_kernel<<<dim3(n / 256), blk, 0, stream>>>(Wq,  wq_bf,  n);
        f32_to_bf16_kernel<<<dim3(2 * n / 256), blk, 0, stream>>>(Wkv, wkv_bf, 2 * n);
        f32_to_bf16_kernel<<<dim3(n / 256), blk, 0, stream>>>(Wo,  wo_bf,  n);
        f32_to_bf16_kernel<<<dim3(n / 256), blk, 0, stream>>>(Wm1, wm1_bf, n);
        f32_to_bf16_kernel<<<dim3(n / 256), blk, 0, stream>>>(Wm2, wm2_bf, n);
    }

    // 2) LayerNorms (with swapaxes + [1:] slice folded in)
    ln_kernel<true ><<<dim3(Bn * S), blk, 0, stream>>>(content, ln_w, ln_b, zc_f32, zc_bf, S, Bn);
    ln_kernel<false><<<dim3(Bn * S), blk, 0, stream>>>(style,   ln_w, ln_b, nullptr, zs_bf, S, Bn);

    // 3) Q = z @ Wq^T + bq            (bf16 out)
    gemm_bf16<true, EP_BF16><<<dim3(D / 128, S / 128, Bn), blk, 0, stream>>>(
        zc_bf, (long long)S * D, D,  wq_bf, 0, D,
        q_bf, (long long)S * D, D,   bq, nullptr, 0, 0, D);

    // 4) KV = zs @ Wkv^T + bkv        (bf16 out, row width 2048: K | V)
    gemm_bf16<true, EP_BF16><<<dim3(N2 / 128, S / 128, Bn), blk, 0, stream>>>(
        zs_bf, (long long)S * D, D,  wkv_bf, 0, D,
        kv_bf, (long long)S * N2, N2, bkv, nullptr, 0, 0, D);

    // 5) scores = Q @ K^T             (fp32 out; K rows are kv rows, ld 2048)
    gemm_bf16<true, EP_F32><<<dim3(N2 / 128, S / 128, Bn), blk, 0, stream>>>(
        q_bf, (long long)S * D, D,   kv_bf, (long long)S * N2, N2,
        sc_f32, (long long)S * N2, N2, nullptr, nullptr, 0, 0, D);

    // 6) softmax(scores / 32) -> bf16 probabilities
    softmax_kernel<<<dim3(Bn * S), blk, 0, stream>>>(sc_f32, probs_bf);

    // 7) PV = P @ V                   (B non-transposed; V = kv + 1024 offset)
    gemm_bf16<false, EP_BF16><<<dim3(D / 128, S / 128, Bn), blk, 0, stream>>>(
        probs_bf, (long long)S * N2, N2,  kv_bf + D, (long long)S * N2, N2,
        pv_bf, (long long)S * D, D,       nullptr, nullptr, 0, 0, N2);

    // 8) xa = PV @ Wo^T + bo          (fp32 out)
    gemm_bf16<true, EP_F32><<<dim3(D / 128, S / 128, Bn), blk, 0, stream>>>(
        pv_bf, (long long)S * D, D,  wo_bf, 0, D,
        xa_f32, (long long)S * D, D, bo, nullptr, 0, 0, D);

    // 9) zxa = LN(z + xa), in-place fp32 + bf16 copy
    resln_kernel<<<dim3(Bn * S), blk, 0, stream>>>(zc_f32, xa_f32, ln_w, ln_b, zxa_bf);

    // 10) h1 = gelu(zxa @ Wm1^T + bm1)  (bf16 out)
    gemm_bf16<true, EP_GELU><<<dim3(D / 128, S / 128, Bn), blk, 0, stream>>>(
        zxa_bf, (long long)S * D, D,  wm1_bf, 0, D,
        h1_bf, (long long)S * D, D,   bm1, nullptr, 0, 0, D);

    // 11) out[s][b][d] = zxa + h1 @ Wm2^T + bm2   (swapaxes folded into strides)
    gemm_bf16<true, EP_FINAL><<<dim3(D / 128, S / 128, Bn), blk, 0, stream>>>(
        h1_bf, (long long)S * D, D,  wm2_bf, 0, D,
        (float*)d_out, (long long)D, Bn * D,
        bm2, zc_f32, (long long)S * D, D, D);

    // 12) second output: _style = style[1:]
    hipMemcpyAsync((float*)d_out + BSD, style + (size_t)Bn * D,
                   BSD * sizeof(float), hipMemcpyDeviceToDevice, stream);
}
